// CustomLSTM_67448166416919
// MI455X (gfx1250) — matrix operand order repaired and checked by reading, not compiled, not run on hardware
//
#include <hip/hip_runtime.h>

// CDNA5 / gfx1250 2-layer LSTM: WMMA f16->f32, weights resident in VGPR
// B-fragments, branchless staging relayout through per-wave LDS,
// c-state in D-fragment registers, software-pipelined x loads.

typedef __attribute__((ext_vector_type(16))) _Float16 v16h;
typedef __attribute__((ext_vector_type(8)))  float    v8f;
typedef float v4f __attribute__((ext_vector_type(4), aligned(4)));

#define T_STEPS 100
#define NF 13
#define NH 13
#define NC 4
#define BATCH_N 32768
#define WAVES_PER_BLOCK 8
#define TILE_M 16

// ---- fast activations: single-instruction TRANS ops where available ----
__device__ __forceinline__ float fast_exp2(float x) {
#if __has_builtin(__builtin_amdgcn_exp2f)
  return __builtin_amdgcn_exp2f(x);
#else
  return __expf(x * 0.6931471805599453f);
#endif
}
__device__ __forceinline__ float fast_rcp(float x) {
#if __has_builtin(__builtin_amdgcn_rcpf)
  return __builtin_amdgcn_rcpf(x);
#else
  return 1.0f / x;
#endif
}
#if __has_builtin(__builtin_amdgcn_tanhf)
__device__ __forceinline__ float tanh_f(float x) {
  return __builtin_amdgcn_tanhf(x);
}
__device__ __forceinline__ float sigmoid_f(float x) {
  return __builtin_amdgcn_tanhf(0.5f * x) * 0.5f + 0.5f;
}
#else
__device__ __forceinline__ float sigmoid_f(float x) {
  // 1/(1+exp(-x)) = rcp(1 + exp2(-log2e * x))
  return fast_rcp(1.0f + fast_exp2(-1.4426950408889634f * x));
}
__device__ __forceinline__ float tanh_f(float x) {
  // tanh(x) = 1 - 2/(exp2(2*log2e*x)+1); saturates correctly at +-1
  return 1.0f - 2.0f * fast_rcp(fast_exp2(2.8853900817779268f * x) + 1.0f);
}
#endif

__device__ __forceinline__ v8f wmma16(v16h a, v16h b, v8f c) {
  // D = A(16x32 f16) x B(32x16 f16) + C(16x16 f32)
  return __builtin_amdgcn_wmma_f32_16x16x32_f16(false, a, false, b,
                                                (short)0, c, false, false);
}
__device__ __forceinline__ void lds_fence() {
  // per-wave DS store->load ordering + compiler barrier
  asm volatile("s_wait_dscnt 0x0" ::: "memory");
}

// Staging layout per wave:
//   A1: 16 rows x 32 f16.  cols [0,16)  = relu(x@Wh+bh)   (valid 0..12)
//                          cols [16,32) = h1              (valid 16..28)
//   H2: 16 rows x 16 f16.  cols [0,16)  = h2              (valid 0..12)
// a2 gathers its "input" halves straight from A1's recurrent block (h1),
// so h1 is stored exactly once per step. Pad columns hold harmless finite
// values; the matching weight B-rows are zero.

__global__ __launch_bounds__(256, 1) void
lstm2_wmma_gfx1250(const float* __restrict__ x,
                   const float* __restrict__ w_hidden,
                   const float* __restrict__ b_hidden,
                   const float* __restrict__ w_lstm1,
                   const float* __restrict__ b_lstm1,
                   const float* __restrict__ w_lstm2,
                   const float* __restrict__ b_lstm2,
                   const float* __restrict__ w_output,
                   const float* __restrict__ b_output,
                   float* __restrict__ out) {
  __shared__ _Float16 sA1[WAVES_PER_BLOCK][TILE_M * 32];
  __shared__ _Float16 sH2[WAVES_PER_BLOCK][TILE_M * 16];

  const int lane = threadIdx.x & 31;
  const int wave = threadIdx.x >> 5;
  const int n    = lane & 15;   // B/C/D column within 16-wide tile
  const int hi   = lane >> 4;   // lane half
  const int m    = n;           // A row (= lane%16)
  const int row0 = hi * 8;      // first C/D row this lane half owns

  _Float16* A1 = sA1[wave];
  _Float16* H2 = sH2[wave];

  for (int i = lane; i < TILE_M * 32; i += 32) A1[i] = (_Float16)0.0f;
  for (int i = lane; i < TILE_M * 16; i += 32) H2[i] = (_Float16)0.0f;
  lds_fence();

  // ---- Resident weight B-fragments (32x16 f16): lane half h -> K = 8*hi+h (h<8), 16+8*hi+(h-8) (h>=8).
  // LSTM W rows: K<13 -> input row K; 16<=K<29 -> recurrent row 13+(K-16).
  v16h wbh, wbo, wb1[4], wb2[4];
#pragma unroll
  for (int h = 0; h < 16; ++h) {
    const int k = (h < 8) ? (hi * 8 + h) : (16 + hi * 8 + (h - 8));
    const bool kin  = (k < NF);                 // input-block row
    const bool krec = (k >= 16 && k < 16 + NH); // recurrent-block row
    const int  rw   = kin ? k : (k - 3);        // 16..28 -> 13..25
    wbh[h] = (kin && n < NH) ? (_Float16)w_hidden[k * NH + n] : (_Float16)0.0f;
    wbo[h] = (kin && n < NC) ? (_Float16)w_output[k * NC + n] : (_Float16)0.0f;
#pragma unroll
    for (int t = 0; t < 4; ++t) {
      const bool v = (kin || krec) && (n < NH);
      wb1[t][h] = v ? (_Float16)w_lstm1[rw * (4 * NH) + t * NH + n]
                    : (_Float16)0.0f;
      wb2[t][h] = v ? (_Float16)w_lstm2[rw * (4 * NH) + t * NH + n]
                    : (_Float16)0.0f;
    }
  }

  // Per-lane scalar biases (column n). FORGET_BIAS folded into f-gate.
  const float bp = (n < NH) ? b_hidden[n] : 0.0f;
  const float bo = (n < NC) ? b_output[n] : 0.0f;
  float b1[4], b2[4];
#pragma unroll
  for (int t = 0; t < 4; ++t) {
    b1[t] = (n < NH) ? b_lstm1[t * NH + n] : 0.0f;
    b2[t] = (n < NH) ? b_lstm2[t * NH + n] : 0.0f;
  }
  b1[2] += 1.0f;
  b2[2] += 1.0f;

  v8f zeroC, c1, c2;
#pragma unroll
  for (int r = 0; r < 8; ++r) { zeroC[r] = 0.0f; c1[r] = 0.0f; c2[r] = 0.0f; }

  // Loop-invariant LDS pointers (stores fold into ds offsets: r*64B / r*32B).
  _Float16*       pIn  = A1 + row0 * 32 + n;        // proj store base
  _Float16*       pRec = A1 + row0 * 32 + 16 + n;   // h1 store base
  _Float16*       pH2  = H2 + row0 * 16 + n;        // h2 store base
  const _Float16* g1   = A1 + m * 32 + hi * 8;      // a1 gather base
  const _Float16* g2a  = A1 + m * 32 + 16 + hi * 8; // a2 lower (h1) base
  const _Float16* g2b  = H2 + m * 16 + hi * 8;      // a2 upper (h2) base

  const long batchBase = (long)(blockIdx.x * WAVES_PER_BLOCK + wave) * TILE_M;
  const float* xrow = x + (batchBase + m) * (long)(T_STEPS * NF);

  // x A-fragment loads (branchless, software-pipelined):
  //   q0 = x[hi*8 .. hi*8+3], q1 = x[4+hi*5 .. 7+hi*5]
  //   hi=0: halves 0..7 = k0..7 ; hi=1: halves 0..4 = k8..12, rest 0.
  const int q1off = 4 + hi * 5;
  v4f q0 = *(const v4f*)(xrow + hi * 8);
  v4f q1 = *(const v4f*)(xrow + q1off);

  for (int t = 0; t < T_STEPS; ++t) {
    // -- assemble x_t fragment from pipelined registers --
    v16h ax;
    ax[0] = (_Float16)q0[0];
    ax[1] = (_Float16)q0[1];
    ax[2] = (_Float16)q0[2];
    ax[3] = (_Float16)q0[3];
    ax[4] = (_Float16)(hi ? q1[3] : q1[0]);
    ax[5] = (_Float16)(hi ? 0.0f : q1[1]);
    ax[6] = (_Float16)(hi ? 0.0f : q1[2]);
    ax[7] = (_Float16)(hi ? 0.0f : q1[3]);
#pragma unroll
    for (int h = 8; h < 16; ++h) ax[h] = (_Float16)0.0f;

    // -- issue next step's x loads early (latency hidden by both layers) --
    const int tn = (t + 1 < T_STEPS) ? (t + 1) : t;
    const float* xpn = xrow + tn * NF;
    q0 = *(const v4f*)(xpn + hi * 8);
    q1 = *(const v4f*)(xpn + q1off);

    // -- input projection + ReLU -> A1 input block (all lanes, branchless) --
    v8f dp = wmma16(ax, wbh, zeroC);
#pragma unroll
    for (int r = 0; r < 8; ++r) {
      float pv = dp[r] + bp;
      pv = pv > 0.0f ? pv : 0.0f;
      pIn[r * 32] = (_Float16)pv;
    }
    lds_fence();

    // -- gather A1 = [proj_t | h1_prev] : 2x ds_load_b128 --
    v16h a1;
#pragma unroll
    for (int h = 0; h < 8; ++h) a1[h] = g1[h];
#pragma unroll
    for (int h = 8; h < 16; ++h) a1[h] = g1[h + 8];

    // -- LSTM1 gates (i, j, f, o = one 16-wide N tile each) --
    v8f zi1 = wmma16(a1, wb1[0], zeroC);
    v8f zj1 = wmma16(a1, wb1[1], zeroC);
    v8f zf1 = wmma16(a1, wb1[2], zeroC);
    v8f zo1 = wmma16(a1, wb1[3], zeroC);
#pragma unroll
    for (int r = 0; r < 8; ++r) {
      const float nc = c1[r] * sigmoid_f(zf1[r] + b1[2]) +
                       sigmoid_f(zi1[r] + b1[0]) * tanh_f(zj1[r] + b1[1]);
      c1[r] = nc;
      pRec[r * 32] = (_Float16)(tanh_f(nc) * sigmoid_f(zo1[r] + b1[3]));
    }
    lds_fence();

    // -- gather A2 = [h1_new (from A1 recurrent) | h2_prev (from H2)] --
    v16h a2;
#pragma unroll
    for (int h = 0; h < 8; ++h) a2[h] = g2a[h];
#pragma unroll
    for (int h = 8; h < 16; ++h) a2[h] = g2b[h - 8];

    // -- LSTM2 gates --
    v8f zi2 = wmma16(a2, wb2[0], zeroC);
    v8f zj2 = wmma16(a2, wb2[1], zeroC);
    v8f zf2 = wmma16(a2, wb2[2], zeroC);
    v8f zo2 = wmma16(a2, wb2[3], zeroC);
#pragma unroll
    for (int r = 0; r < 8; ++r) {
      const float nc = c2[r] * sigmoid_f(zf2[r] + b2[2]) +
                       sigmoid_f(zi2[r] + b2[0]) * tanh_f(zj2[r] + b2[1]);
      c2[r] = nc;
      pH2[r * 16] = (_Float16)(tanh_f(nc) * sigmoid_f(zo2[r] + b2[3]));
    }
  }
  lds_fence();

  // -- output head: h2 gather from H2 (upper K-halves are zero rows) --
  v16h ah2;
#pragma unroll
  for (int h = 0; h < 8; ++h) ah2[h] = g2b[h];
#pragma unroll
  for (int h = 8; h < 16; ++h) ah2[h] = (_Float16)0.0f;
  v8f dout = wmma16(ah2, wbo, zeroC);
  if (n < NC) {
#pragma unroll
    for (int r = 0; r < 8; ++r) {
      out[(batchBase + row0 + r) * NC + n] = dout[r] + bo;
    }
  }
}

extern "C" void kernel_launch(void* const* d_in, const int* in_sizes, int n_in,
                              void* d_out, int out_size, void* d_ws,
                              size_t ws_size, hipStream_t stream) {
  (void)in_sizes; (void)n_in; (void)d_ws; (void)ws_size; (void)out_size;
  const float* x        = (const float*)d_in[0];
  const float* w_hidden = (const float*)d_in[1];
  const float* b_hidden = (const float*)d_in[2];
  const float* w_lstm1  = (const float*)d_in[3];
  const float* b_lstm1  = (const float*)d_in[4];
  const float* w_lstm2  = (const float*)d_in[5];
  const float* b_lstm2  = (const float*)d_in[6];
  const float* w_output = (const float*)d_in[7];
  const float* b_output = (const float*)d_in[8];
  float* out = (float*)d_out;

  const int rows_per_block = WAVES_PER_BLOCK * TILE_M;  // 128
  dim3 grid(BATCH_N / rows_per_block);                  // 256 blocks
  dim3 block(256);                                      // 8 waves
  lstm2_wmma_gfx1250<<<grid, block, 0, stream>>>(
      x, w_hidden, b_hidden, w_lstm1, b_lstm1, w_lstm2, b_lstm2,
      w_output, b_output, out);
}
